// MultiScaleGambaLayer_6030134083929
// MI455X (gfx1250) — matrix operand order repaired and checked by reading, not compiled, not run on hardware
//
#include <hip/hip_runtime.h>
#include <stdint.h>

typedef __attribute__((ext_vector_type(16))) __bf16 v16bf;
typedef __attribute__((ext_vector_type(8)))  __bf16 v8bf;
typedef __attribute__((ext_vector_type(8)))  float  v8f;

#define HDIM 128

__device__ __forceinline__ v16bf cat16(v8bf lo, v8bf hi) {
  return __builtin_shufflevector(lo, hi, 0,1,2,3,4,5,6,7,8,9,10,11,12,13,14,15);
}

// A-fragment (16x32 bf16, row-major source with leading dim ld).
// lane<16 holds row=lane, k = 0..15;
// lane>=16 holds row=lane-16, k = 16..31. Two 16B loads per lane.
__device__ __forceinline__ v16bf ldA(const __bf16* tile, int ld, int lane) {
  const __bf16* p = tile + (size_t)(lane & 15) * ld + ((lane >> 4) << 4);
  v8bf lo = *(const v8bf*)p;
  v8bf hi = *(const v8bf*)(p + 8);
  return cat16(lo, hi);
}

// B-fragment from pre-swizzled weights: [kt][nt][lane][16] halves, 32B/lane.
__device__ __forceinline__ v16bf ldBsw(const __bf16* w, int kt, int nt, int NT, int lane) {
  const __bf16* p = w + (((((size_t)kt * NT + nt) * 32) + lane) << 4);
  v8bf lo = *(const v8bf*)p;
  v8bf hi = *(const v8bf*)(p + 8);
  return cat16(lo, hi);
}

__device__ __forceinline__ v8f wmma_bf16(v16bf a, v16bf b, v8f c) {
  return __builtin_amdgcn_wmma_f32_16x16x32_bf16(false, a, false, b, (short)0, c, false, false);
}

__device__ __forceinline__ float silu_f(float v) { return v / (1.f + expf(-v)); }
__device__ __forceinline__ float softplus_f(float v) { return (v > 20.f) ? v : log1pf(expf(v)); }

// ---------------- generic utility kernels ----------------
__global__ void k_zero(uint32_t* p, int n) {
  int i = blockIdx.x * 256 + threadIdx.x;
  if (i < n) p[i] = 0u;
}
__global__ void k_copy(float* dst, const float* src, int n) {
  int i = blockIdx.x * 256 + threadIdx.x;
  if (i < n) dst[i] = src[i];
}
// Cast fp32 KxN weight -> bf16 in WMMA B-fragment-swizzled order.
__global__ void k_swz(const float* W, __bf16* out, int K, int Nn) {
  int idx = blockIdx.x * 256 + threadIdx.x;
  if (idx >= K * Nn) return;
  int k = idx / Nn, n = idx - k * Nn;
  int kt = k >> 5, kr = k & 31, nt = n >> 4;
  int lane = (kr & 16) + (n & 15);
  int j = kr & 15;
  int NT = Nn >> 4;
  out[(((((size_t)kt * NT + nt) * 32) + lane) << 4) + j] = (__bf16)W[idx];
}

// ---------------- edge message passing ----------------
// Block = 256 threads (8 waves). One 16-edge tile per block iteration:
// A = concat(pe[dst], pe[src], edge_attr) 16x384 bf16 in LDS, h1 = A@W1+b1,
// LN+ReLU, msg = h2@W2+b2, atomic scatter-add into agg[dst].
__global__ void __launch_bounds__(256, 1) k_edge(
    const float* __restrict__ pe, const int* __restrict__ ei, const float* __restrict__ ea,
    const __bf16* __restrict__ w1sw, const float* __restrict__ b1,
    const float* __restrict__ lng, const float* __restrict__ lnb,
    const __bf16* __restrict__ w2sw, const float* __restrict__ b2,
    float* __restrict__ agg, int E)
{
  __shared__ __align__(16) __bf16 sA[16 * 384];
  __shared__ __align__(16) float  sH[16 * 128];
  __shared__ __align__(16) __bf16 sA2[16 * 128];
  __shared__ float sMu[16], sRs[16];
  __shared__ int   sDst[16];
  const int wave = threadIdx.x >> 5, lane = threadIdx.x & 31;
  const int ntiles = (E + 15) >> 4;
  for (int tile = blockIdx.x; tile < ntiles; tile += gridDim.x) {
    const int e0 = tile << 4;
    // stage A tile (gather + fp32->bf16)
    for (int idx = threadIdx.x; idx < 16 * 384; idx += 256) {
      int r = idx / 384, c = idx - r * 384;
      int e = e0 + r;
      float v = 0.f;
      int d = -1;
      if (e < E) {
        int s = ei[e];
        d = ei[E + e];
        if (c < 128)      v = pe[(size_t)d * HDIM + c];
        else if (c < 256) v = pe[(size_t)s * HDIM + (c - 128)];
        else              v = ea[(size_t)e * HDIM + (c - 256)];
      }
      sA[idx] = (__bf16)v;
      if (c == 0) sDst[r] = d;
    }
    __syncthreads();
    // matmul1: wave owns n-tile `wave` (16 cols), K=384 (12 k-steps)
    v8f acc = {};
    #pragma unroll
    for (int kt = 0; kt < 12; ++kt)
      acc = wmma_bf16(ldA(sA + kt * 32, 384, lane), ldBsw(w1sw, kt, wave, 8, lane), acc);
    {
      int n = (wave << 4) + (lane & 15);
      float bias = b1[n];
      int mb = (lane >> 4) << 3;
      #pragma unroll
      for (int r = 0; r < 8; ++r) sH[(mb + r) * 128 + n] = acc[r] + bias;
    }
    __syncthreads();
    if (threadIdx.x < 16) {
      int r = threadIdx.x;
      float mu = 0.f;
      for (int c = 0; c < 128; ++c) mu += sH[r * 128 + c];
      mu *= (1.f / 128.f);
      float var = 0.f;
      for (int c = 0; c < 128; ++c) { float d = sH[r * 128 + c] - mu; var += d * d; }
      sMu[r] = mu; sRs[r] = rsqrtf(var * (1.f / 128.f) + 1e-5f);
    }
    __syncthreads();
    for (int idx = threadIdx.x; idx < 16 * 128; idx += 256) {
      int r = idx >> 7, c = idx & 127;
      float h = (sH[idx] - sMu[r]) * sRs[r] * lng[c] + lnb[c];
      sA2[idx] = (__bf16)fmaxf(h, 0.f);
    }
    __syncthreads();
    // matmul2: K=128 (4 k-steps)
    v8f acc2 = {};
    #pragma unroll
    for (int kt = 0; kt < 4; ++kt)
      acc2 = wmma_bf16(ldA(sA2 + kt * 32, 128, lane), ldBsw(w2sw, kt, wave, 8, lane), acc2);
    {
      int n = (wave << 4) + (lane & 15);
      float bias = b2[n];
      int mb = (lane >> 4) << 3;
      #pragma unroll
      for (int r = 0; r < 8; ++r) {
        int d = sDst[mb + r];
        if (d >= 0) unsafeAtomicAdd(&agg[(size_t)d * HDIM + n], acc2[r] + bias);
      }
    }
    __syncthreads();
  }
}

// pe = LN(pe + agg): one wave per row, float4 per lane, shfl reductions.
__global__ void __launch_bounds__(256) k_pe_ln(
    float* __restrict__ pe, const float* __restrict__ agg,
    const float* __restrict__ g, const float* __restrict__ bb, int n)
{
  int row = blockIdx.x * 8 + (threadIdx.x >> 5);
  int lane = threadIdx.x & 31;
  if (row >= n) return;
  float4 v = ((const float4*)(pe + (size_t)row * 128))[lane];
  float4 a = ((const float4*)(agg + (size_t)row * 128))[lane];
  float x0 = v.x + a.x, x1 = v.y + a.y, x2 = v.z + a.z, x3 = v.w + a.w;
  float s = x0 + x1 + x2 + x3;
  for (int o = 16; o; o >>= 1) s += __shfl_xor(s, o, 32);
  float mu = s * (1.f / 128.f);
  float d0 = x0 - mu, d1 = x1 - mu, d2 = x2 - mu, d3 = x3 - mu;
  float vs = d0 * d0 + d1 * d1 + d2 * d2 + d3 * d3;
  for (int o = 16; o; o >>= 1) vs += __shfl_xor(vs, o, 32);
  float rs = rsqrtf(vs * (1.f / 128.f) + 1e-5f);
  int c = lane << 2;
  float4 go = *(const float4*)(g + c);
  float4 bo = *(const float4*)(bb + c);
  float4 out = { d0 * rs * go.x + bo.x, d1 * rs * go.y + bo.y,
                 d2 * rs * go.z + bo.z, d3 * rs * go.w + bo.w };
  ((float4*)(pe + (size_t)row * 128))[lane] = out;
}

// ---------------- dense / mamba branch ----------------
__global__ void k_first_init(int* first) { if (threadIdx.x < 32) first[threadIdx.x] = 0x7fffffff; }
__global__ void k_first_min(const int* batch, int* first, int n) {
  int i = blockIdx.x * 256 + threadIdx.x;
  if (i < n) atomicMin(&first[batch[i]], i);
}
// scatter x into dense (B,4096,128) bf16, row-major + B-fragment-swizzled copies
__global__ void k_scatter(const float* __restrict__ x, const int* __restrict__ batch,
                          const int* __restrict__ first, __bf16* __restrict__ xd,
                          __bf16* __restrict__ xsw, int n)
{
  int i = blockIdx.x, c = threadIdx.x;
  if (i >= n) return;
  int b = batch[i];
  int pos = i - first[b];
  if (pos >= 4096) return;
  __bf16 bv = (__bf16)x[(size_t)i * 128 + c];
  xd[((size_t)(b << 12) + pos) * 128 + c] = bv;
  int kt = pos >> 5, kr = pos & 31, nt = c >> 4;
  int lane = (kr & 16) + (c & 15);
  int j = kr & 15;
  xsw[((((((size_t)b * 128 + kt) * 8) + nt) * 32 + lane) << 4) + j] = bv;
}

// T[b,n,k] = X[b] @ theta : wave-per-mtile, WMMA
__global__ void __launch_bounds__(256) k_T(const __bf16* __restrict__ xd,
                                           const __bf16* __restrict__ thsw,
                                           float* __restrict__ T)
{
  int lane = threadIdx.x & 31;
  int wid = blockIdx.x * 8 + (threadIdx.x >> 5);
  int nw = gridDim.x * 8;
  for (int mt = wid; mt < 32 * 256; mt += nw) {
    int b = mt >> 8, m = (mt & 255) << 4;
    const __bf16* abase = xd + ((size_t)(b << 12) + m) * 128;
    for (int nt = 0; nt < 2; ++nt) {
      v8f acc = {};
      #pragma unroll
      for (int kt = 0; kt < 4; ++kt)
        acc = wmma_bf16(ldA(abase + kt * 32, 128, lane), ldBsw(thsw, kt, nt, 2, lane), acc);
      int ncol = (nt << 4) + (lane & 15);
      int mb = (lane >> 4) << 3;
      #pragma unroll
      for (int r = 0; r < 8; ++r)
        T[((size_t)(b << 12) + m + mb + r) * 32 + ncol] = acc[r];
    }
  }
}

// AX[b] = T[b]^T @ X[b] : block per (b, m-tile of 2), K=4096 via LDS-staged A tiles
__global__ void __launch_bounds__(256) k_AX(const float* __restrict__ T,
                                            const __bf16* __restrict__ xsw,
                                            float* __restrict__ ax)
{
  __shared__ __align__(16) __bf16 sTA[16 * 32];
  int b = blockIdx.x >> 1, mt = blockIdx.x & 1;
  int wave = threadIdx.x >> 5, lane = threadIdx.x & 31;
  const __bf16* xb = xsw + (size_t)b * 128 * 8 * 32 * 16;
  v8f acc = {};
  for (int kt = 0; kt < 128; ++kt) {
    int k0 = kt << 5;
    for (int idx = threadIdx.x; idx < 512; idx += 256) {
      int m = idx >> 5, kk = idx & 31;   // A(m,k) = T[b][k][mt*16+m]
      sTA[m * 32 + kk] = (__bf16)T[((size_t)(b << 12) + k0 + kk) * 32 + (mt << 4) + m];
    }
    __syncthreads();
    acc = wmma_bf16(ldA(sTA, 32, lane), ldBsw(xb, kt, wave, 8, lane), acc);
    __syncthreads();
  }
  int ncol = (wave << 4) + (lane & 15);
  int mb = (lane >> 4) << 3;
  #pragma unroll
  for (int r = 0; r < 8; ++r)
    ax[((size_t)b * 32 + (mt << 4) + mb + r) * 128 + ncol] = acc[r];
}

// xz = ax @ m_in_w  (1024x128 @ 128x512)
__global__ void k_xz(const float* __restrict__ ax, const float* __restrict__ w, float* __restrict__ xz) {
  int idx = blockIdx.x * 256 + threadIdx.x;
  if (idx >= 1024 * 512) return;
  int bt = idx >> 9, j = idx & 511;
  float s = 0.f;
  for (int k = 0; k < 128; ++k) s += ax[bt * 128 + k] * w[k * 512 + j];
  xz[idx] = s;
}
// causal depthwise conv (DCONV=4) + silu
__global__ void k_conv(const float* __restrict__ xz, const float* __restrict__ cw,
                       const float* __restrict__ cb, float* __restrict__ uc) {
  int idx = blockIdx.x * 256 + threadIdx.x;
  if (idx >= 1024 * 256) return;
  int bt = idx >> 8, d = idx & 255;
  int b = bt >> 5, t = bt & 31;
  float s = cb[d];
  #pragma unroll
  for (int k = 0; k < 4; ++k) {
    int tt = t + k - 3;
    if (tt >= 0) s += xz[((b << 5) + tt) * 512 + d] * cw[d * 4 + k];
  }
  uc[idx] = silu_f(s);
}
// xdb = uc @ m_xproj_w (1024x256 @ 256x264)
__global__ void k_xdb(const float* __restrict__ uc, const float* __restrict__ w, float* __restrict__ xdb) {
  int idx = blockIdx.x * 256 + threadIdx.x;
  if (idx >= 1024 * 264) return;
  int bt = idx / 264, j = idx - bt * 264;
  float s = 0.f;
  for (int d = 0; d < 256; ++d) s += uc[bt * 256 + d] * w[d * 264 + j];
  xdb[idx] = s;
}
// dt = softplus(xdb[:, :8] @ m_dt_w + b)
__global__ void k_dt(const float* __restrict__ xdb, const float* __restrict__ w,
                     const float* __restrict__ b, float* __restrict__ dt) {
  int idx = blockIdx.x * 256 + threadIdx.x;
  if (idx >= 1024 * 256) return;
  int bt = idx >> 8, d = idx & 255;
  float s = b[d];
  #pragma unroll
  for (int r = 0; r < 8; ++r) s += xdb[bt * 264 + r] * w[r * 256 + d];
  dt[idx] = softplus_f(s);
}
// selective scan; only final-step y is needed. block=(128 s-lanes), grid=B*DIN
__global__ void __launch_bounds__(128) k_scan(const float* __restrict__ dt, const float* __restrict__ xdb,
                                              const float* __restrict__ uc, const float* __restrict__ Alog,
                                              float* __restrict__ ylast) {
  int b = blockIdx.x >> 8, d = blockIdx.x & 255, s = threadIdx.x;
  float A = -expf(Alog[d * 128 + s]);
  float h = 0.f;
  for (int t = 0; t < 32; ++t) {
    int bt = (b << 5) + t;
    float dtt = dt[bt * 256 + d];
    float Bs = xdb[bt * 264 + 8 + s];
    float uv = uc[bt * 256 + d];
    h = expf(dtt * A) * h + dtt * Bs * uv;
  }
  __shared__ float red[128];
  red[s] = h * xdb[((b << 5) + 31) * 264 + 136 + s];
  __syncthreads();
  for (int o = 64; o > 0; o >>= 1) { if (s < o) red[s] += red[s + o]; __syncthreads(); }
  if (s == 0) ylast[(b << 8) + d] = red[0];
}
__global__ void k_yl2(const float* __restrict__ ylast, const float* __restrict__ uc,
                      const float* __restrict__ xz, const float* __restrict__ mD,
                      float* __restrict__ yl2) {
  int idx = blockIdx.x * 256 + threadIdx.x;
  if (idx >= 32 * 256) return;
  int b = idx >> 8, d = idx & 255;
  int bt = (b << 5) + 31;
  float y = ylast[idx] + mD[d] * uc[bt * 256 + d];
  yl2[idx] = y * silu_f(xz[bt * 512 + 256 + d]);
}
__global__ void k_xm(const float* __restrict__ yl2, const float* __restrict__ w, float* __restrict__ xm) {
  int idx = blockIdx.x * 256 + threadIdx.x;
  if (idx >= 32 * 128) return;
  int b = idx >> 7, hh = idx & 127;
  float s = 0.f;
  for (int d = 0; d < 256; ++d) s += yl2[b * 256 + d] * w[d * 128 + hh];
  xm[idx] = s;
}

// ---------------- final fused MLP + residual LN ----------------
__global__ void __launch_bounds__(256, 1) k_final(
    const float* __restrict__ pe, const float* __restrict__ xm, const int* __restrict__ batch,
    const float* __restrict__ x,
    const __bf16* __restrict__ w1sw, const float* __restrict__ b1,
    const float* __restrict__ lng1, const float* __restrict__ lnb1,
    const __bf16* __restrict__ w2sw, const float* __restrict__ b2,
    const float* __restrict__ lng2, const float* __restrict__ lnb2,
    float* __restrict__ out, int n)
{
  __shared__ __align__(16) __bf16 sA[16 * 256];
  __shared__ __align__(16) float  sH[16 * 128];
  __shared__ __align__(16) __bf16 sA2[16 * 128];
  __shared__ float sMu[16], sRs[16];
  const int wave = threadIdx.x >> 5, lane = threadIdx.x & 31;
  const int ntiles = (n + 15) >> 4;
  for (int tile = blockIdx.x; tile < ntiles; tile += gridDim.x) {
    const int i0 = tile << 4;
    for (int idx = threadIdx.x; idx < 16 * 256; idx += 256) {
      int r = idx >> 8, c = idx & 255;
      int i = i0 + r;
      float v = 0.f;
      if (i < n) v = (c < 128) ? pe[(size_t)i * 128 + c]
                               : xm[(size_t)batch[i] * 128 + (c - 128)];
      sA[idx] = (__bf16)v;
    }
    __syncthreads();
    v8f acc = {};
    #pragma unroll
    for (int kt = 0; kt < 8; ++kt)
      acc = wmma_bf16(ldA(sA + kt * 32, 256, lane), ldBsw(w1sw, kt, wave, 8, lane), acc);
    {
      int ncol = (wave << 4) + (lane & 15);
      float bias = b1[ncol];
      int mb = (lane >> 4) << 3;
      #pragma unroll
      for (int r = 0; r < 8; ++r) sH[(mb + r) * 128 + ncol] = acc[r] + bias;
    }
    __syncthreads();
    if (threadIdx.x < 16) {
      int r = threadIdx.x;
      float mu = 0.f;
      for (int c = 0; c < 128; ++c) mu += sH[r * 128 + c];
      mu *= (1.f / 128.f);
      float var = 0.f;
      for (int c = 0; c < 128; ++c) { float d = sH[r * 128 + c] - mu; var += d * d; }
      sMu[r] = mu; sRs[r] = rsqrtf(var * (1.f / 128.f) + 1e-5f);
    }
    __syncthreads();
    for (int idx = threadIdx.x; idx < 16 * 128; idx += 256) {
      int r = idx >> 7, c = idx & 127;
      float h = (sH[idx] - sMu[r]) * sRs[r] * lng1[c] + lnb1[c];
      sA2[idx] = (__bf16)fmaxf(h, 0.f);
    }
    __syncthreads();
    v8f acc2 = {};
    #pragma unroll
    for (int kt = 0; kt < 4; ++kt)
      acc2 = wmma_bf16(ldA(sA2 + kt * 32, 128, lane), ldBsw(w2sw, kt, wave, 8, lane), acc2);
    __syncthreads();
    {
      int ncol = (wave << 4) + (lane & 15);
      float bias = b2[ncol];
      int mb = (lane >> 4) << 3;
      #pragma unroll
      for (int r = 0; r < 8; ++r) {
        int i = i0 + mb + r;
        float xv = (i < n) ? x[(size_t)i * 128 + ncol] : 0.f;
        sH[(mb + r) * 128 + ncol] = acc2[r] + bias + xv;   // residual
      }
    }
    __syncthreads();
    if (threadIdx.x < 16) {
      int r = threadIdx.x;
      float mu = 0.f;
      for (int c = 0; c < 128; ++c) mu += sH[r * 128 + c];
      mu *= (1.f / 128.f);
      float var = 0.f;
      for (int c = 0; c < 128; ++c) { float d = sH[r * 128 + c] - mu; var += d * d; }
      sMu[r] = mu; sRs[r] = rsqrtf(var * (1.f / 128.f) + 1e-5f);
    }
    __syncthreads();
    for (int idx = threadIdx.x; idx < 16 * 128; idx += 256) {
      int r = idx >> 7, c = idx & 127;
      int i = i0 + r;
      if (i < n) out[(size_t)i * 128 + c] = (sH[idx] - sMu[r]) * sRs[r] * lng2[c] + lnb2[c];
    }
    __syncthreads();
  }
}

// ---------------- host launch ----------------
extern "C" void kernel_launch(void* const* d_in, const int* in_sizes, int n_in,
                              void* d_out, int out_size, void* d_ws, size_t ws_size,
                              hipStream_t stream) {
  const float* x        = (const float*)d_in[0];
  const int*   ei       = (const int*)d_in[1];
  const float* ea       = (const float*)d_in[2];
  const int*   batch    = (const int*)d_in[3];
  const float* sc_w1    = (const float*)d_in[4];
  const float* sc_b1    = (const float*)d_in[5];
  const float* sc_ln_g  = (const float*)d_in[6];
  const float* sc_ln_b  = (const float*)d_in[7];
  const float* sc_w2    = (const float*)d_in[8];
  const float* sc_b2    = (const float*)d_in[9];
  const float* sc_fn_g  = (const float*)d_in[10];
  const float* sc_fn_b  = (const float*)d_in[11];
  const float* theta_w  = (const float*)d_in[12];
  const float* m_in_w   = (const float*)d_in[13];
  const float* m_conv_w = (const float*)d_in[14];
  const float* m_conv_b = (const float*)d_in[15];
  const float* m_xproj_w= (const float*)d_in[16];
  const float* m_dt_w   = (const float*)d_in[17];
  const float* m_dt_b   = (const float*)d_in[18];
  const float* m_Alog   = (const float*)d_in[19];
  const float* m_D      = (const float*)d_in[20];
  const float* m_out_w  = (const float*)d_in[21];
  const float* mg_w1    = (const float*)d_in[22];
  const float* mg_b1    = (const float*)d_in[23];
  const float* mg_ln_g  = (const float*)d_in[24];
  const float* mg_ln_b  = (const float*)d_in[25];
  const float* mg_w2    = (const float*)d_in[26];
  const float* mg_b2    = (const float*)d_in[27];
  const float* ln_g     = (const float*)d_in[28];
  const float* ln_b     = (const float*)d_in[29];

  const int N = in_sizes[0] / 128;
  const int E = in_sizes[1] / 2;
  (void)n_in; (void)ws_size;

  char* ws = (char*)d_ws;
  size_t off = 0;
  auto alloc = [&](size_t bytes) -> char* {
    off = (off + 255) & ~(size_t)255;
    char* p = ws + off;
    off += bytes;
    return p;
  };
  float*  pe    = (float*)alloc((size_t)N * 128 * 4);
  float*  agg   = (float*)alloc((size_t)N * 128 * 4);
  float*  T     = (float*)alloc((size_t)32 * 4096 * 32 * 4);
  float*  xz    = (float*)alloc((size_t)1024 * 512 * 4);
  float*  uc    = (float*)alloc((size_t)1024 * 256 * 4);
  float*  xdb   = (float*)alloc((size_t)1024 * 264 * 4);
  float*  dtv   = (float*)alloc((size_t)1024 * 256 * 4);
  float*  ax    = (float*)alloc((size_t)32 * 32 * 128 * 4);
  float*  ylast = (float*)alloc((size_t)32 * 256 * 4);
  float*  yl2   = (float*)alloc((size_t)32 * 256 * 4);
  float*  xm    = (float*)alloc((size_t)32 * 128 * 4);
  int*    first = (int*)alloc(32 * 4);
  __bf16* xd    = (__bf16*)alloc((size_t)32 * 4096 * 128 * 2);
  __bf16* xsw   = (__bf16*)alloc((size_t)32 * 4096 * 128 * 2);
  __bf16* w1sw  = (__bf16*)alloc((size_t)384 * 128 * 2);
  __bf16* w2sw  = (__bf16*)alloc((size_t)128 * 128 * 2);
  __bf16* thsw  = (__bf16*)alloc((size_t)128 * 32 * 2);
  __bf16* g1sw  = (__bf16*)alloc((size_t)256 * 128 * 2);
  __bf16* g2sw  = (__bf16*)alloc((size_t)128 * 128 * 2);

  // weight casts / swizzles
  k_swz<<<(384 * 128 + 255) / 256, 256, 0, stream>>>(sc_w1, w1sw, 384, 128);
  k_swz<<<(128 * 128 + 255) / 256, 256, 0, stream>>>(sc_w2, w2sw, 128, 128);
  k_swz<<<(128 * 32 + 255) / 256, 256, 0, stream>>>(theta_w, thsw, 128, 32);
  k_swz<<<(256 * 128 + 255) / 256, 256, 0, stream>>>(mg_w1, g1sw, 256, 128);
  k_swz<<<(128 * 128 + 255) / 256, 256, 0, stream>>>(mg_w2, g2sw, 128, 128);

  // pe = x
  k_copy<<<((size_t)N * 128 + 255) / 256, 256, 0, stream>>>(pe, x, N * 128);

  // 4 message-passing iterations
  for (int it = 0; it < 4; ++it) {
    k_zero<<<((size_t)N * 128 + 255) / 256, 256, 0, stream>>>((uint32_t*)agg, N * 128);
    k_edge<<<1024, 256, 0, stream>>>(pe, ei, ea, w1sw, sc_b1, sc_ln_g, sc_ln_b,
                                     w2sw, sc_b2, agg, E);
    k_pe_ln<<<(N + 7) / 8, 256, 0, stream>>>(pe, agg, sc_fn_g, sc_fn_b, N);
  }

  // dense scatter
  k_first_init<<<1, 32, 0, stream>>>(first);
  k_first_min<<<(N + 255) / 256, 256, 0, stream>>>(batch, first, N);
  {
    int nd = (int)((size_t)32 * 4096 * 128 * 2 / 4);
    k_zero<<<(nd + 255) / 256, 256, 0, stream>>>((uint32_t*)xd, nd);
    k_zero<<<(nd + 255) / 256, 256, 0, stream>>>((uint32_t*)xsw, nd);
  }
  k_scatter<<<N, 128, 0, stream>>>(x, batch, first, xd, xsw, N);

  // alpha / ax (WMMA)
  k_T<<<256, 256, 0, stream>>>(xd, thsw, T);
  k_AX<<<64, 256, 0, stream>>>(T, xsw, ax);

  // mamba
  k_xz<<<(1024 * 512 + 255) / 256, 256, 0, stream>>>(ax, m_in_w, xz);
  k_conv<<<(1024 * 256 + 255) / 256, 256, 0, stream>>>(xz, m_conv_w, m_conv_b, uc);
  k_xdb<<<(1024 * 264 + 255) / 256, 256, 0, stream>>>(uc, m_xproj_w, xdb);
  k_dt<<<(1024 * 256 + 255) / 256, 256, 0, stream>>>(xdb, m_dt_w, m_dt_b, dtv);
  k_scan<<<32 * 256, 128, 0, stream>>>(dtv, xdb, uc, m_Alog, ylast);
  k_yl2<<<(32 * 256 + 255) / 256, 256, 0, stream>>>(ylast, uc, xz, m_D, yl2);
  k_xm<<<(32 * 128 + 255) / 256, 256, 0, stream>>>(yl2, m_out_w, xm);

  // final fused MLP + residual LN -> d_out
  k_final<<<1024, 256, 0, stream>>>(pe, xm, batch, x, g1sw, mg_b1, mg_ln_g, mg_ln_b,
                                    g2sw, mg_b2, ln_g, ln_b, (float*)d_out, N);
}
